// CrossModalAttention_75316546503126
// MI455X (gfx1250) — compile-verified, matrix operand order checked by reading, not run on hardware
//
#include <hip/hip_runtime.h>
#include <hip/hip_bf16.h>

typedef __attribute__((ext_vector_type(16))) __bf16 v16bf;
typedef __attribute__((ext_vector_type(8)))  __bf16 v8bf;
typedef __attribute__((ext_vector_type(8)))  float  v8f;

#define BATCH 8
#define SEQ   2048
#define DIM   768
#define DAUD  16
#define M_TOT (BATCH * SEQ)

#define BM 128
#define BN 128
#define BK 32
#define NK (DIM / BK)        // 24 k-steps
#define LDSROW 40            // 32 bf16 + 8 pad; row stride = 80B (16B aligned)

// ---------------------------------------------------------------------------
// Kernel 1: v_row[b,:] = ((acoustic[b,:] @ Wa) + ba) @ Wv + bv    (tiny)
// ---------------------------------------------------------------------------
__global__ __launch_bounds__(256)
void cm_vrow_kernel(const float* __restrict__ ac, const float* __restrict__ Wa,
                    const float* __restrict__ ba, const float* __restrict__ Wv,
                    const float* __restrict__ bv, float* __restrict__ vrow) {
    __shared__ float fa[DIM];
    const int b   = blockIdx.x;
    const int tid = threadIdx.x;

    for (int j = tid; j < DIM; j += 256) {
        float acc = ba[j];
#pragma unroll
        for (int k = 0; k < DAUD; ++k)
            acc += ac[b * DAUD + k] * Wa[k * DIM + j];
        fa[j] = acc;
    }
    __syncthreads();
    for (int j = tid; j < DIM; j += 256) {
        float acc = bv[j];
        for (int k = 0; k < DIM; ++k)
            acc += fa[k] * Wv[k * DIM + j];
        vrow[b * DIM + j] = acc;
    }
}

// ---------------------------------------------------------------------------
// Kernel 2: out = text @ Wt + bt + broadcast(v_row)
//   bf16 WMMA, f32 accum, double-buffered LDS, packed b32 transpose stores
// ---------------------------------------------------------------------------
__global__ __launch_bounds__(256)
void cm_gemm_kernel(const float* __restrict__ A,     // [M_TOT][DIM] text
                    const float* __restrict__ W,     // [DIM][DIM]   Wt (k-major)
                    const float* __restrict__ bias,  // [DIM]        bt
                    const float* __restrict__ vrow,  // [BATCH][DIM]
                    float* __restrict__ out) {       // [M_TOT][DIM]
    __shared__ __bf16 As[2][BM][LDSROW];   // As[buf][m][k]
    __shared__ __bf16 Bs[2][BN][LDSROW];   // Bs[buf][n][k]  (transposed W tile)

    const int tid  = threadIdx.x;
    const int wave = tid >> 5;
    const int lane = tid & 31;
    const int khi  = lane >> 4;   // which K half-group this lane holds
    const int lm   = lane & 15;   // m (A frag) or n (B frag) within tile

    const int m0 = blockIdx.y * BM;
    const int n0 = blockIdx.x * BN;

    // A staging map: 32 rows/pass x 8 thread-cols x 4 floats, 4 passes
    const int a_tr = tid >> 3;          // 0..31
    const int a_tc = (tid & 7) * 4;     // 0,4,...,28
    // B staging map: thread -> k-pair (2*kp, 2*kp+1) x 8 columns
    const int b_kp = tid & 15;          // k pair index -> k = 2*kp
    const int b_ng = tid >> 4;          // column group -> n = ng*8 .. +7

    v8f zero = {};
    v8f acc[BN / 16];
#pragma unroll
    for (int i = 0; i < BN / 16; ++i) acc[i] = zero;

    float4 ra[4];         // A staging regs
    float4 rb0[2], rb1[2];// B staging regs: rows 2kp and 2kp+1, 8 floats each

    // ---- prologue: load tile 0 -> regs -> LDS[0] ----
    {
        const int k0 = 0;
#pragma unroll
        for (int p = 0; p < 4; ++p)
            ra[p] = *(const float4*)(&A[(size_t)(m0 + a_tr + p * 32) * DIM + k0 + a_tc]);
#pragma unroll
        for (int p = 0; p < 2; ++p) {
            rb0[p] = *(const float4*)(&W[(size_t)(k0 + 2 * b_kp)     * DIM + n0 + b_ng * 8 + p * 4]);
            rb1[p] = *(const float4*)(&W[(size_t)(k0 + 2 * b_kp + 1) * DIM + n0 + b_ng * 8 + p * 4]);
        }
#pragma unroll
        for (int p = 0; p < 4; ++p) {
            const int row = a_tr + p * 32;
            As[0][row][a_tc + 0] = (__bf16)ra[p].x;
            As[0][row][a_tc + 1] = (__bf16)ra[p].y;
            As[0][row][a_tc + 2] = (__bf16)ra[p].z;
            As[0][row][a_tc + 3] = (__bf16)ra[p].w;
        }
#pragma unroll
        for (int p = 0; p < 2; ++p) {
            const float* lo = (const float*)&rb0[p];
            const float* hi = (const float*)&rb1[p];
#pragma unroll
            for (int c = 0; c < 4; ++c) {
                union { __bf16 h[2]; unsigned int u; } pk;
                pk.h[0] = (__bf16)lo[c];
                pk.h[1] = (__bf16)hi[c];
                *(unsigned int*)(&Bs[0][b_ng * 8 + p * 4 + c][2 * b_kp]) = pk.u;
            }
        }
    }
    __syncthreads();

    for (int t = 0; t < NK; ++t) {
        const int cur = t & 1;

        // ---- issue next tile's global loads (hide latency under WMMAs) ----
        if (t + 1 < NK) {
            const int k0 = (t + 1) * BK;
#pragma unroll
            for (int p = 0; p < 4; ++p)
                ra[p] = *(const float4*)(&A[(size_t)(m0 + a_tr + p * 32) * DIM + k0 + a_tc]);
#pragma unroll
            for (int p = 0; p < 2; ++p) {
                rb0[p] = *(const float4*)(&W[(size_t)(k0 + 2 * b_kp)     * DIM + n0 + b_ng * 8 + p * 4]);
                rb1[p] = *(const float4*)(&W[(size_t)(k0 + 2 * b_kp + 1) * DIM + n0 + b_ng * 8 + p * 4]);
            }
        }
        // ---- prefetch two tiles ahead on the big A stream ----
        if (t + 2 < NK) {
            const int k2 = (t + 2) * BK;
#pragma unroll
            for (int p = 0; p < 4; ++p)
                __builtin_prefetch(&A[(size_t)(m0 + a_tr + p * 32) * DIM + k2 + a_tc], 0, 1);
        }

        // ---- compute from LDS[cur] ----
        v16bf afrag;
        {
            const __bf16* base = &As[cur][wave * 16 + lm][0];
            ((v8bf*)&afrag)[0] = *(const v8bf*)(base + khi * 8);        // K = khi*8 .. +7
            ((v8bf*)&afrag)[1] = *(const v8bf*)(base + 16 + khi * 8);   // K = 16+khi*8 .. +7
        }
#pragma unroll
        for (int nt = 0; nt < BN / 16; ++nt) {
            v16bf bfrag;
            const __bf16* base = &Bs[cur][nt * 16 + lm][0];
            ((v8bf*)&bfrag)[0] = *(const v8bf*)(base + khi * 8);
            ((v8bf*)&bfrag)[1] = *(const v8bf*)(base + 16 + khi * 8);
            acc[nt] = __builtin_amdgcn_wmma_f32_16x16x32_bf16(
                false, afrag, false, bfrag, (short)0, acc[nt], false, false);
        }

        // ---- convert + store next tile into LDS[1-cur] ----
        if (t + 1 < NK) {
            const int nxt = 1 - cur;
#pragma unroll
            for (int p = 0; p < 4; ++p) {
                const int row = a_tr + p * 32;
                As[nxt][row][a_tc + 0] = (__bf16)ra[p].x;
                As[nxt][row][a_tc + 1] = (__bf16)ra[p].y;
                As[nxt][row][a_tc + 2] = (__bf16)ra[p].z;
                As[nxt][row][a_tc + 3] = (__bf16)ra[p].w;
            }
#pragma unroll
            for (int p = 0; p < 2; ++p) {
                const float* lo = (const float*)&rb0[p];
                const float* hi = (const float*)&rb1[p];
#pragma unroll
                for (int c = 0; c < 4; ++c) {
                    union { __bf16 h[2]; unsigned int u; } pk;
                    pk.h[0] = (__bf16)lo[c];
                    pk.h[1] = (__bf16)hi[c];
                    *(unsigned int*)(&Bs[nxt][b_ng * 8 + p * 4 + c][2 * b_kp]) = pk.u;
                }
            }
        }
        __syncthreads();
    }

    // ---- epilogue: C layout lane l, vgpr r -> m = khi*8+r, n = lm ----
    const int row_base = m0 + wave * 16 + khi * 8;
    const int b = row_base / SEQ;   // 16-row aligned tiles; SEQ % 16 == 0
#pragma unroll
    for (int nt = 0; nt < BN / 16; ++nt) {
        const int col  = n0 + nt * 16 + lm;
        const float ad = bias[col] + vrow[b * DIM + col];
#pragma unroll
        for (int r = 0; r < 8; ++r)
            out[(size_t)(row_base + r) * DIM + col] = acc[nt][r] + ad;
    }
}

// ---------------------------------------------------------------------------
extern "C" void kernel_launch(void* const* d_in, const int* in_sizes, int n_in,
                              void* d_out, int out_size, void* d_ws, size_t ws_size,
                              hipStream_t stream) {
    const float* text = (const float*)d_in[0];   // [B,S,768]
    const float* ac   = (const float*)d_in[1];   // [B,16]
    const float* Wt   = (const float*)d_in[2];   // [768,768]
    const float* bt   = (const float*)d_in[3];   // [768]
    const float* Wa   = (const float*)d_in[4];   // [16,768]
    const float* ba   = (const float*)d_in[5];   // [768]
    // Wq/bq/Wk/bk (6..9) cancel mathematically (uniform softmax over equal scores)
    const float* Wv   = (const float*)d_in[10];  // [768,768]
    const float* bv   = (const float*)d_in[11];  // [768]
    float* out        = (float*)d_out;

    float* vrow = (float*)d_ws;                  // [B,768] scratch (24 KB)

    cm_vrow_kernel<<<dim3(BATCH), dim3(256), 0, stream>>>(ac, Wa, ba, Wv, bv, vrow);

    dim3 grid(DIM / BN, M_TOT / BM);             // (6, 128)
    cm_gemm_kernel<<<grid, dim3(256), 0, stream>>>(text, Wt, bt, vrow, out);
}
